// FluxSingleTransformerBlock_88089779241100
// MI455X (gfx1250) — matrix operand order repaired and checked by reading, not compiled, not run on hardware
//
#include <hip/hip_runtime.h>
#include <hip/hip_bf16.h>
#include <math.h>

// ---------------- problem constants ----------------
#define DIMC   3072
#define HEADS  24
#define HDIM   128
#define MLPC   12288
#define BB     2
#define TEXT   256
#define IMG    1024
#define SS     (TEXT + IMG)       // 1280
#define CATC   (DIMC + MLPC)      // 15360
#define EPSV   1e-6f

// ---------------- WMMA types ----------------
typedef __attribute__((ext_vector_type(16))) __bf16 v16bf;
typedef __attribute__((ext_vector_type(8)))  float  v8f;

union FragBF {
  v16bf v;
  uint4 u[2];
};

// ---------------- async LDS copy (gfx1250) ----------------
#if __has_builtin(__builtin_amdgcn_global_load_async_to_lds_b128)
#define HAS_ASYNC_LDS 1
typedef int v4i_t __attribute__((ext_vector_type(4)));
typedef __attribute__((address_space(1))) v4i_t gv4i_t;
typedef __attribute__((address_space(3))) v4i_t lv4i_t;
#define ASYNC_LD_B128(gp, lp, off) \
  __builtin_amdgcn_global_load_async_to_lds_b128((gv4i_t*)(gp), (lv4i_t*)(lp), (off), 0)
#define ASYNC_WAIT() asm volatile("s_wait_asynccnt 0" ::: "memory")
#else
#define HAS_ASYNC_LDS 0
#endif

__device__ __forceinline__ v8f wmma_bf16(const FragBF& a, const FragBF& b, v8f c) {
  return __builtin_amdgcn_wmma_f32_16x16x32_bf16(false, a.v, false, b.v, (short)0, c,
                                                 false, false);
}

__device__ __forceinline__ unsigned short f2bf(float f) {
  unsigned int u = __float_as_uint(f);
  u += 0x7FFFu + ((u >> 16) & 1u);          // round-to-nearest-even
  return (unsigned short)(u >> 16);
}
__device__ __forceinline__ unsigned int pack_bf2(float lo, float hi) {
  return (unsigned int)f2bf(lo) | ((unsigned int)f2bf(hi) << 16);
}

__device__ __forceinline__ float siluf(float x) { return x / (1.0f + __expf(-x)); }
__device__ __forceinline__ float geluf(float x) {
  return 0.5f * x * (1.0f + tanhf(0.7978845608028654f * (x + 0.044715f * x * x * x)));
}

// =====================================================================
// 1) silu(temb)
// =====================================================================
__global__ void silu_temb_kernel(const float* __restrict__ temb, float* __restrict__ out) {
  int i = blockIdx.x * blockDim.x + threadIdx.x;
  if (i < BB * DIMC) out[i] = siluf(temb[i]);
}

// =====================================================================
// 2) cond_in = silu(temb + mean_s(text_h))
// =====================================================================
__global__ void textmean_kernel(const float* __restrict__ hidden,
                                const float* __restrict__ temb,
                                float* __restrict__ cond_in) {
  int c = blockIdx.x * blockDim.x + threadIdx.x;   // 0..DIMC-1
  int b = blockIdx.y;
  if (c >= DIMC) return;
  float s = 0.f;
  for (int t = 0; t < TEXT; ++t)
    s += hidden[((size_t)b * SS + t) * DIMC + c];
  s = s * (1.0f / TEXT) + temb[b * DIMC + c];
  cond_in[b * DIMC + c] = siluf(s);
}

// =====================================================================
// 3) small matrix-vector: out[b][n] = act[b][:] . W[:,n] + bias[n]  (K = DIMC)
// =====================================================================
__global__ void small_mv_kernel(const float* __restrict__ act,
                                const float* __restrict__ W,
                                const float* __restrict__ bias,
                                float* __restrict__ out, int N) {
  __shared__ float av[DIMC];
  int b = blockIdx.y;
  for (int i = threadIdx.x; i < DIMC; i += blockDim.x) av[i] = act[b * DIMC + i];
  __syncthreads();
  int n = blockIdx.x * blockDim.x + threadIdx.x;
  if (n >= N) return;
  float s = 0.f;
  for (int k = 0; k < DIMC; ++k) s += av[k] * W[(size_t)k * N + n];
  out[(size_t)b * N + n] = s + bias[n];
}

// =====================================================================
// 4) LayerNorm + adaLN modulation -> bf16 activations
// =====================================================================
__global__ void ln_mod_kernel(const float* __restrict__ hidden,
                              const float* __restrict__ e,   // [B][3*DIMC]
                              unsigned short* __restrict__ xbf) {
  int row = blockIdx.x;                 // 0 .. B*S-1
  int b = row / SS;
  const float* hp = hidden + (size_t)row * DIMC;
  float s = 0.f, s2 = 0.f;
  for (int c = threadIdx.x; c < DIMC; c += 256) {
    float v = hp[c];
    s += v; s2 += v * v;
  }
  for (int off = 16; off; off >>= 1) { s += __shfl_xor(s, off); s2 += __shfl_xor(s2, off); }
  __shared__ float rs[8], rs2[8];
  int wave = threadIdx.x >> 5;
  if ((threadIdx.x & 31) == 0) { rs[wave] = s; rs2[wave] = s2; }
  __syncthreads();
  __shared__ float mu_s, rinv_s;
  if (threadIdx.x == 0) {
    float ts = 0.f, ts2 = 0.f;
    for (int i = 0; i < 8; ++i) { ts += rs[i]; ts2 += rs2[i]; }
    float mu = ts / DIMC;
    float var = ts2 / DIMC - mu * mu;
    mu_s = mu; rinv_s = rsqrtf(var + EPSV);
  }
  __syncthreads();
  float mu = mu_s, rinv = rinv_s;
  const float* shift = e + (size_t)b * 3 * DIMC;
  const float* scale = shift + DIMC;
  unsigned short* op = xbf + (size_t)row * DIMC;
  for (int c = threadIdx.x; c < DIMC; c += 256) {
    float v = (hp[c] - mu) * rinv;
    v = v * (1.0f + scale[c]) + shift[c];
    op[c] = f2bf(v);
  }
}

// =====================================================================
// 5) Tiled WMMA GEMM:  C[M,N] = A_bf16[M,K] * W_f32[K,N] + bias
//    128x128x32 tiles, 8 waves / block.
//    A tile: GLOBAL_LOAD_ASYNC_TO_LDS_B128 (bf16 pass-through).
//    W tile: fp32 -> bf16 convert + transpose, packed b32 LDS stores.
// =====================================================================
#define BM 128
#define BN 128
#define BK 32
#define LDA (BK + 8)

__global__ void __launch_bounds__(256)
gemm_bf16w_kernel(const unsigned short* __restrict__ A,
                  const float* __restrict__ W,
                  const float* __restrict__ bias,
                  float* __restrict__ C, int M, int N, int K) {
  __shared__ unsigned short As[BM][LDA];
  __shared__ unsigned short Bs[BN][LDA];
  const int tid = threadIdx.x;
  const int lane = tid & 31;
  const int wave = tid >> 5;
  const int waveM = wave >> 2;     // 0..1 -> 64 rows each
  const int waveN = wave & 3;      // 0..3 -> 32 cols each
  const int r16 = lane & 15;
  const int half = lane >> 4;
  const int m0 = blockIdx.y * BM;
  const int n0 = blockIdx.x * BN;

  v8f zf = {};
  v8f acc[4][2];
#pragma unroll
  for (int i = 0; i < 4; ++i)
#pragma unroll
    for (int j = 0; j < 2; ++j) acc[i][j] = zf;

  const int arow = tid >> 1, ach = (tid & 1) * 16;       // A: 16 bf16 per thread
  const int krow = (tid >> 4) * 2, nch = (tid & 15) * 8; // W: 2 k-rows x 8 n per thread

  for (int k0 = 0; k0 < K; k0 += BK) {
    // --- stage A tile (bf16, async byte copy to LDS) ---
    const unsigned short* ap = &A[(size_t)(m0 + arow) * K + k0 + ach];
#if HAS_ASYNC_LDS
    ASYNC_LD_B128(ap, &As[arow][ach], 0);
    ASYNC_LD_B128(ap, &As[arow][ach], 16);
#else
    *(uint4*)&As[arow][ach] = ((const uint4*)ap)[0];
    *(uint4*)&As[arow][ach + 8] = ((const uint4*)ap)[1];
#endif
    // --- stage W tile: transpose + fp32->bf16, packed pair stores ---
    const float4* w0 = (const float4*)&W[(size_t)(k0 + krow) * N + n0 + nch];
    const float4* w1 = (const float4*)&W[(size_t)(k0 + krow + 1) * N + n0 + nch];
    float4 a0 = w0[0], a1 = w0[1];
    float4 b0 = w1[0], b1 = w1[1];
    *(unsigned int*)&Bs[nch + 0][krow] = pack_bf2(a0.x, b0.x);
    *(unsigned int*)&Bs[nch + 1][krow] = pack_bf2(a0.y, b0.y);
    *(unsigned int*)&Bs[nch + 2][krow] = pack_bf2(a0.z, b0.z);
    *(unsigned int*)&Bs[nch + 3][krow] = pack_bf2(a0.w, b0.w);
    *(unsigned int*)&Bs[nch + 4][krow] = pack_bf2(a1.x, b1.x);
    *(unsigned int*)&Bs[nch + 5][krow] = pack_bf2(a1.y, b1.y);
    *(unsigned int*)&Bs[nch + 6][krow] = pack_bf2(a1.z, b1.z);
    *(unsigned int*)&Bs[nch + 7][krow] = pack_bf2(a1.w, b1.w);
    // prefetch next fp32 weight tile into cache hierarchy
    if (k0 + BK < K) {
      __builtin_prefetch(&W[(size_t)(k0 + BK + krow) * N + n0 + nch], 0, 1);
      __builtin_prefetch(&W[(size_t)(k0 + BK + krow + 1) * N + n0 + nch], 0, 1);
    }
#if HAS_ASYNC_LDS
    ASYNC_WAIT();
#endif
    __syncthreads();

    FragBF af[4];
#pragma unroll
    for (int mt = 0; mt < 4; ++mt) {
      int row = waveM * 64 + mt * 16 + r16;
      af[mt].u[0] = *(const uint4*)&As[row][half * 8];
      af[mt].u[1] = *(const uint4*)&As[row][16 + half * 8];
    }
    FragBF bf[2];
#pragma unroll
    for (int nt = 0; nt < 2; ++nt) {
      int nrow = waveN * 32 + nt * 16 + r16;
      bf[nt].u[0] = *(const uint4*)&Bs[nrow][half * 8];
      bf[nt].u[1] = *(const uint4*)&Bs[nrow][16 + half * 8];
    }
#pragma unroll
    for (int mt = 0; mt < 4; ++mt)
#pragma unroll
      for (int nt = 0; nt < 2; ++nt) acc[mt][nt] = wmma_bf16(af[mt], bf[nt], acc[mt][nt]);
    __syncthreads();
  }

#pragma unroll
  for (int mt = 0; mt < 4; ++mt)
#pragma unroll
    for (int nt = 0; nt < 2; ++nt) {
      int col = n0 + waveN * 32 + nt * 16 + r16;
      float bv = bias[col];
#pragma unroll
      for (int j = 0; j < 8; ++j) {
        int row = m0 + waveM * 64 + mt * 16 + half * 8 + j;
        C[(size_t)row * N + col] = acc[mt][nt][j] + bv;
      }
    }
}

// =====================================================================
// 6) Q/K post: per-head RMSNorm (gamma) + RoPE -> bf16  [b][h][s][d]
// =====================================================================
__global__ void __launch_bounds__(128)
qk_post_kernel(const float* __restrict__ t,       // [B][S][DIMC]
               const float* __restrict__ g,       // [HDIM]
               const float* __restrict__ cosb,    // [S][HDIM/2]
               const float* __restrict__ sinb,
               unsigned short* __restrict__ obf)  // [B][H][S][HDIM]
{
  int id = blockIdx.x;
  int h = id % HEADS;
  int s = (id / HEADS) % SS;
  int b = id / (HEADS * SS);
  int d = threadIdx.x;
  float v = t[((size_t)b * SS + s) * DIMC + h * HDIM + d];
  float ss = v * v;
  for (int off = 16; off; off >>= 1) ss += __shfl_xor(ss, off);
  __shared__ float red[4];
  if ((threadIdx.x & 31) == 0) red[threadIdx.x >> 5] = ss;
  __syncthreads();
  float tot = red[0] + red[1] + red[2] + red[3];
  float vn = v * rsqrtf(tot * (1.0f / HDIM) + EPSV) * g[d];
  __shared__ float buf[HDIM];
  buf[d] = vn;
  __syncthreads();
  int i = d >> 1;
  float c = cosb[(size_t)s * (HDIM / 2) + i];
  float sn = sinb[(size_t)s * (HDIM / 2) + i];
  float r;
  if ((d & 1) == 0) r = vn * c - buf[d + 1] * sn;
  else              r = buf[d - 1] * sn + vn * c;
  obf[((size_t)(b * HEADS + h) * SS + s) * HDIM + d] = f2bf(r);
}

// =====================================================================
// 7) V pack: fp32 [b][s][h*128+d] -> bf16 transposed [b][h][d][s]
// =====================================================================
__global__ void vpack_kernel(const float* __restrict__ v, unsigned short* __restrict__ vt) {
  size_t idx = (size_t)blockIdx.x * blockDim.x + threadIdx.x;
  if (idx >= (size_t)BB * SS * DIMC) return;
  int c = idx % DIMC;
  int s = (idx / DIMC) % SS;
  int b = idx / ((size_t)SS * DIMC);
  int h = c / HDIM, d = c % HDIM;
  vt[((size_t)(b * HEADS + h) * HDIM + d) * SS + s] = f2bf(v[idx]);
}

// =====================================================================
// 8) Flash-style attention. grid (S/64, HEADS, B), 4 waves (128 thr).
//    Each wave owns a 16-row q stripe; O = 16x128 fp32 accum in VGPRs.
// =====================================================================
__global__ void __launch_bounds__(128)
attn_kernel(const unsigned short* __restrict__ qbf,   // [b][h][s][d]
            const unsigned short* __restrict__ kbf,   // [b][h][s][d]
            const unsigned short* __restrict__ vt,    // [b][h][d][s]
            float* __restrict__ attn_out)             // [b][s][DIMC]
{
  __shared__ unsigned short qs[64][HDIM + 8];
  __shared__ unsigned short ps[4][16][64 + 8];
  const int b = blockIdx.z, hd = blockIdx.y, qt = blockIdx.x;
  const int tid = threadIdx.x, lane = tid & 31, wave = tid >> 5;
  const int r16 = lane & 15, half = lane >> 4;
  const size_t qbase = ((size_t)(b * HEADS + hd) * SS + qt * 64) * HDIM;
  const size_t kbase = ((size_t)(b * HEADS + hd) * SS) * HDIM;
  const size_t vbase = ((size_t)(b * HEADS + hd) * HDIM) * SS;

  // load 64x128 q tile (async byte copy)
  for (int i = tid; i < 64 * 16; i += 128) {
    int row = i >> 4, c8 = (i & 15) * 8;
#if HAS_ASYNC_LDS
    ASYNC_LD_B128(&qbf[qbase + (size_t)row * HDIM + c8], &qs[row][c8], 0);
#else
    *(uint4*)&qs[row][c8] = *(const uint4*)&qbf[qbase + (size_t)row * HDIM + c8];
#endif
  }
#if HAS_ASYNC_LDS
  ASYNC_WAIT();
#endif
  __syncthreads();

  v8f zf = {};
  v8f o[8];
#pragma unroll
  for (int i = 0; i < 8; ++i) o[i] = zf;
  float mrow[8], lrow[8];
#pragma unroll
  for (int j = 0; j < 8; ++j) { mrow[j] = -1e30f; lrow[j] = 0.f; }
  const float scl = 0.08838834764831845f;  // 1/sqrt(128)

  for (int kt = 0; kt < SS / 64; ++kt) {
    v8f sacc[4];
#pragma unroll
    for (int nt = 0; nt < 4; ++nt) sacc[nt] = zf;

    // S = Q K^T over d
#pragma unroll
    for (int ks = 0; ks < 4; ++ks) {
      FragBF a;
      int qrow = wave * 16 + r16;
      a.u[0] = *(const uint4*)&qs[qrow][ks * 32 + half * 8];
      a.u[1] = *(const uint4*)&qs[qrow][ks * 32 + 16 + half * 8];
#pragma unroll
      for (int nt = 0; nt < 4; ++nt) {
        int key = kt * 64 + nt * 16 + r16;
        const uint4* kp = (const uint4*)&kbf[kbase + (size_t)key * HDIM + ks * 32 + half * 8];
        FragBF bf; bf.u[0] = kp[0]; bf.u[1] = kp[2];
        sacc[nt] = wmma_bf16(a, bf, sacc[nt]);
      }
    }
#pragma unroll
    for (int nt = 0; nt < 4; ++nt)
#pragma unroll
      for (int j = 0; j < 8; ++j) sacc[nt][j] *= scl;

    // online softmax (rows live in j, cols across 16 lanes x 4 tiles)
    float cmax[8];
#pragma unroll
    for (int j = 0; j < 8; ++j) {
      float m = sacc[0][j];
      m = fmaxf(m, sacc[1][j]); m = fmaxf(m, sacc[2][j]); m = fmaxf(m, sacc[3][j]);
      for (int off = 1; off < 16; off <<= 1) m = fmaxf(m, __shfl_xor(m, off));
      cmax[j] = m;
    }
#pragma unroll
    for (int j = 0; j < 8; ++j) {
      float mnew = fmaxf(mrow[j], cmax[j]);
      float alpha = __expf(mrow[j] - mnew);
      float rs = 0.f;
#pragma unroll
      for (int nt = 0; nt < 4; ++nt) {
        float p = __expf(sacc[nt][j] - mnew);
        sacc[nt][j] = p;
        rs += p;
      }
      for (int off = 1; off < 16; off <<= 1) rs += __shfl_xor(rs, off);
      lrow[j] = lrow[j] * alpha + rs;
      mrow[j] = mnew;
#pragma unroll
      for (int dt = 0; dt < 8; ++dt) o[dt][j] *= alpha;
    }

    __syncthreads();   // protect ps reads of previous iteration
#pragma unroll
    for (int nt = 0; nt < 4; ++nt)
#pragma unroll
      for (int j = 0; j < 8; ++j)
        ps[wave][half * 8 + j][nt * 16 + r16] = f2bf(sacc[nt][j]);
    __syncthreads();

    // O += P V
#pragma unroll
    for (int ks2 = 0; ks2 < 2; ++ks2) {
      FragBF a2;
      a2.u[0] = *(const uint4*)&ps[wave][r16][ks2 * 32 + half * 8];
      a2.u[1] = *(const uint4*)&ps[wave][r16][ks2 * 32 + 16 + half * 8];
#pragma unroll
      for (int dt = 0; dt < 8; ++dt) {
        int d = dt * 16 + r16;
        const uint4* vp = (const uint4*)&vt[vbase + (size_t)d * SS + kt * 64 + ks2 * 32 + half * 8];
        FragBF bf; bf.u[0] = vp[0]; bf.u[1] = vp[2];
        o[dt] = wmma_bf16(a2, bf, o[dt]);
      }
    }
  }

  // normalize + write [b][s][hd*128+d]
#pragma unroll
  for (int dt = 0; dt < 8; ++dt) {
    int c = hd * HDIM + dt * 16 + r16;
#pragma unroll
    for (int j = 0; j < 8; ++j) {
      int s = qt * 64 + wave * 16 + half * 8 + j;
      attn_out[((size_t)b * SS + s) * DIMC + c] = o[dt][j] / lrow[j];
    }
  }
}

// =====================================================================
// 9) concat pack: [attn_out | gelu(mlp)] -> bf16 [M][CATC]
// =====================================================================
__global__ void pack_kernel(const float* __restrict__ attn_out,
                            const float* __restrict__ mlp,
                            unsigned short* __restrict__ cat) {
  size_t idx = (size_t)blockIdx.x * blockDim.x + threadIdx.x;
  if (idx >= (size_t)BB * SS * CATC) return;
  size_t m = idx / CATC;
  int c = idx % CATC;
  float v;
  if (c < DIMC) v = attn_out[m * DIMC + c];
  else          v = geluf(mlp[m * MLPC + (c - DIMC)]);
  cat[idx] = f2bf(v);
}

// =====================================================================
// 10) selector mask: mask = ((img_h + cond) . w_sel + b_sel > 0)
// =====================================================================
__global__ void mask_kernel(const float* __restrict__ hidden,
                            const float* __restrict__ cond,
                            const float* __restrict__ w_sel,
                            const float* __restrict__ b_sel,
                            float* __restrict__ mask) {
  int si = blockIdx.x;       // 0..IMG-1
  int b = blockIdx.y;
  const float* hp = hidden + ((size_t)b * SS + TEXT + si) * DIMC;
  const float* cp = cond + (size_t)b * DIMC;
  float s = 0.f;
  for (int c = threadIdx.x; c < DIMC; c += 256) s += (hp[c] + cp[c]) * w_sel[c];
  for (int off = 16; off; off >>= 1) s += __shfl_xor(s, off);
  __shared__ float red[8];
  if ((threadIdx.x & 31) == 0) red[threadIdx.x >> 5] = s;
  __syncthreads();
  if (threadIdx.x == 0) {
    float t = b_sel[0];
    for (int i = 0; i < 8; ++i) t += red[i];
    mask[(size_t)b * IMG + si] = (t > 0.f) ? 1.0f : 0.0f;
  }
}

// =====================================================================
// 11) epilogue: out = residual + gate * y2 * (img ? mask : 1)
// =====================================================================
__global__ void epilogue_kernel(const float* __restrict__ hidden,
                                const float* __restrict__ e,
                                const float* __restrict__ y2,
                                const float* __restrict__ mask,
                                float* __restrict__ out) {
  size_t idx = (size_t)blockIdx.x * blockDim.x + threadIdx.x;
  if (idx >= (size_t)BB * SS * DIMC) return;
  int c = idx % DIMC;
  int s = (idx / DIMC) % SS;
  int b = idx / ((size_t)SS * DIMC);
  float gate = e[(size_t)b * 3 * DIMC + 2 * DIMC + c];
  float mk = (s < TEXT) ? 1.0f : mask[(size_t)b * IMG + (s - TEXT)];
  out[idx] = hidden[idx] + gate * y2[idx] * mk;
}

// =====================================================================
// launch
// =====================================================================
extern "C" void kernel_launch(void* const* d_in, const int* in_sizes, int n_in,
                              void* d_out, int out_size, void* d_ws, size_t ws_size,
                              hipStream_t stream) {
  const float* hidden  = (const float*)d_in[0];
  const float* temb    = (const float*)d_in[1];
  const float* rot_cos = (const float*)d_in[2];
  const float* rot_sin = (const float*)d_in[3];
  const float* w_ada   = (const float*)d_in[4];
  const float* b_ada   = (const float*)d_in[5];
  const float* wq      = (const float*)d_in[6];
  const float* bq      = (const float*)d_in[7];
  const float* wk      = (const float*)d_in[8];
  const float* bk      = (const float*)d_in[9];
  const float* wv      = (const float*)d_in[10];
  const float* bv      = (const float*)d_in[11];
  const float* q_scale = (const float*)d_in[12];
  const float* k_scale = (const float*)d_in[13];
  const float* w_cond  = (const float*)d_in[14];
  const float* b_cond  = (const float*)d_in[15];
  const float* w_sel   = (const float*)d_in[16];
  const float* b_sel   = (const float*)d_in[17];
  const float* w1      = (const float*)d_in[18];
  const float* b1      = (const float*)d_in[19];
  const float* w2      = (const float*)d_in[20];
  const float* b2      = (const float*)d_in[21];
  float* out = (float*)d_out;

  char* ws = (char*)d_ws;
  size_t off = 0;
  auto alloc = [&](size_t bytes) -> char* {
    char* p = ws + off;
    off = (off + bytes + 255) & ~(size_t)255;
    return p;
  };
  const size_t M = (size_t)BB * SS;            // 2560
  float* silu1   = (float*)alloc(BB * DIMC * 4);
  float* cond_in = (float*)alloc(BB * DIMC * 4);
  float* e_buf   = (float*)alloc(BB * 3 * DIMC * 4);
  float* cond    = (float*)alloc(BB * DIMC * 4);
  unsigned short* x_bf = (unsigned short*)alloc(M * DIMC * 2);
  float* qf      = (float*)alloc(M * DIMC * 4);
  float* kf      = (float*)alloc(M * DIMC * 4);
  float* vf      = (float*)alloc(M * DIMC * 4);
  unsigned short* q_bf = (unsigned short*)alloc(M * DIMC * 2);
  unsigned short* k_bf = (unsigned short*)alloc(M * DIMC * 2);
  unsigned short* v_t  = (unsigned short*)alloc(M * DIMC * 2);
  float* attn_o  = (float*)alloc(M * DIMC * 4);
  float* mlp     = (float*)alloc(M * MLPC * 4);
  unsigned short* cat = (unsigned short*)alloc(M * CATC * 2);
  float* y2      = (float*)alloc(M * DIMC * 4);
  float* maskb   = (float*)alloc(BB * IMG * 4);

  // --- small conditioning path ---
  silu_temb_kernel<<<(BB * DIMC + 255) / 256, 256, 0, stream>>>(temb, silu1);
  textmean_kernel<<<dim3(DIMC / 256, BB), 256, 0, stream>>>(hidden, temb, cond_in);
  small_mv_kernel<<<dim3(3 * DIMC / 256, BB), 256, 0, stream>>>(silu1, w_ada, b_ada, e_buf, 3 * DIMC);
  small_mv_kernel<<<dim3(DIMC / 256, BB), 256, 0, stream>>>(cond_in, w_cond, b_cond, cond, DIMC);

  // --- LN + modulation ---
  ln_mod_kernel<<<M, 256, 0, stream>>>(hidden, e_buf, x_bf);

  // --- QKV projections (WMMA) ---
  dim3 gqkv(DIMC / BN, M / BM);
  gemm_bf16w_kernel<<<gqkv, 256, 0, stream>>>(x_bf, wq, bq, qf, (int)M, DIMC, DIMC);
  gemm_bf16w_kernel<<<gqkv, 256, 0, stream>>>(x_bf, wk, bk, kf, (int)M, DIMC, DIMC);
  gemm_bf16w_kernel<<<gqkv, 256, 0, stream>>>(x_bf, wv, bv, vf, (int)M, DIMC, DIMC);

  // --- q/k rmsnorm + rope, v transpose pack ---
  qk_post_kernel<<<BB * SS * HEADS, 128, 0, stream>>>(qf, q_scale, rot_cos, rot_sin, q_bf);
  qk_post_kernel<<<BB * SS * HEADS, 128, 0, stream>>>(kf, k_scale, rot_cos, rot_sin, k_bf);
  vpack_kernel<<<(unsigned)((M * DIMC + 255) / 256), 256, 0, stream>>>(vf, v_t);

  // --- attention (flash-style WMMA) ---
  attn_kernel<<<dim3(SS / 64, HEADS, BB), 128, 0, stream>>>(q_bf, k_bf, v_t, attn_o);

  // --- MLP up (WMMA) ---
  gemm_bf16w_kernel<<<dim3(MLPC / BN, M / BM), 256, 0, stream>>>(x_bf, w1, b1, mlp, (int)M, MLPC, DIMC);

  // --- concat + gelu pack, then fused down-proj (WMMA) ---
  pack_kernel<<<(unsigned)((M * CATC + 255) / 256), 256, 0, stream>>>(attn_o, mlp, cat);
  gemm_bf16w_kernel<<<dim3(DIMC / BN, M / BM), 256, 0, stream>>>(cat, w2, b2, y2, (int)M, DIMC, CATC);

  // --- selector mask + epilogue ---
  mask_kernel<<<dim3(IMG, BB), 256, 0, stream>>>(hidden, cond, w_sel, b_sel, maskb);
  epilogue_kernel<<<(unsigned)((M * DIMC + 255) / 256), 256, 0, stream>>>(hidden, e_buf, y2, maskb, out);
}
